// max_embedding_45354854645902
// MI455X (gfx1250) — matrix operand order repaired and checked by reading, not compiled, not run on hardware
//
#include <hip/hip_runtime.h>
#include <hip/hip_bf16.h>
#include <math.h>

// ---------------------------------------------------------------------------
// Problem constants (match reference setup_inputs)
// ---------------------------------------------------------------------------
#define FEAT 64
#define NBATCH 32
#define H1 256
#define H2 128
#define BN_EPS 1e-5f

typedef __attribute__((ext_vector_type(16))) _Float16 v16h;
typedef __attribute__((ext_vector_type(8)))  float    v8f;

// ---------------------------------------------------------------------------
// Kernel 0: init pooled buffer to -inf (d_ws is poisoned by the harness)
// ---------------------------------------------------------------------------
__global__ void init_pooled_kernel(float* __restrict__ pooled, int n) {
    int i = blockIdx.x * blockDim.x + threadIdx.x;
    if (i < n) pooled[i] = -INFINITY;
}

// ---------------------------------------------------------------------------
// Kernel 1: streaming segment-max over F [N, 64] with sorted batch_idx.
// 256 threads/block = 16 row-groups; each row-group covers one row via
// 16 lanes * float4 = 64 features (global_load_b128, fully coalesced).
// Each thread carries a running max for its current segment; flushes via
// float atomicMax (global_atomic_max_num_f32) only at segment boundaries.
// ---------------------------------------------------------------------------
#define ROWS_PER_BLOCK 2048

__device__ __forceinline__ void flush_max(float* __restrict__ pooled,
                                          int seg, int f4, const float4& m) {
    float* p = pooled + seg * FEAT + f4 * 4;
    atomicMax(p + 0, m.x);
    atomicMax(p + 1, m.y);
    atomicMax(p + 2, m.z);
    atomicMax(p + 3, m.w);
}

__global__ __launch_bounds__(256)
void segmax_kernel(const float* __restrict__ F,
                   const int* __restrict__ batch_idx,
                   float* __restrict__ pooled, int N) {
    const int tid      = threadIdx.x;
    const int rowGroup = tid >> 4;   // 0..15
    const int f4       = tid & 15;   // which float4 of the 64-feature row
    const long base    = (long)blockIdx.x * ROWS_PER_BLOCK;

    float4 cur = make_float4(-INFINITY, -INFINITY, -INFINITY, -INFINITY);
    int curSeg = -1;

    for (int i = rowGroup; i < ROWS_PER_BLOCK; i += 16) {
        const long r = base + i;
        if (r >= N) break;
        // prefetch a future row of this thread's stream (global_prefetch_b8)
        if (r + 64 < N)
            __builtin_prefetch(F + (r + 64) * FEAT + f4 * 4, 0, 0);

        const int seg = batch_idx[r];
        const float4 v =
            *reinterpret_cast<const float4*>(F + r * FEAT + f4 * 4);

        if (seg != curSeg) {                 // sorted idx: rare (<= 32 total)
            if (curSeg >= 0) flush_max(pooled, curSeg, f4, cur);
            curSeg = seg;
            cur = v;
        } else {
            cur.x = fmaxf(cur.x, v.x);
            cur.y = fmaxf(cur.y, v.y);
            cur.z = fmaxf(cur.z, v.z);
            cur.w = fmaxf(cur.w, v.w);
        }
    }
    if (curSeg >= 0) flush_max(pooled, curSeg, f4, cur);
}

// ---------------------------------------------------------------------------
// Kernel 2: fc1 + BatchNorm(train) + ReLU + fc2, single block, 32 waves.
// WMMA tiling (v_wmma_f32_16x16x32_f16):
//   fc1: [32,64]@[64,256]^T -> 2x16 tiles of 16x16, one per wave, K=64 (2 steps)
//   fc2: [32,256]@[256,128]^T -> 2x8 tiles, waves 0..15, K=256 (8 steps)
// A-frag (16x32 f16): lane L holds row L&15; a[i] -> K=(L>>4)*16+i.
// B-frag (32x16 f16): lane L holds col L&15; b[i] = B[(L>>4)*16+i][L&15].
//   With B = W^T this is a contiguous row segment of row-major W.
// D-frag f32: d[v] -> row v + 8*(L>>4), col L&15.
// ---------------------------------------------------------------------------
__device__ __forceinline__ v16h load_a_f16(const _Float16* __restrict__ row,
                                           int half /*lane>>4*/) {
    v16h a;
    const _Float16* p = row + half * 16;
#pragma unroll
    for (int i = 0; i < 16; ++i) {
        a[i]     = p[i];
    }
    return a;
}

__device__ __forceinline__ v16h load_b_from_f32(const float* __restrict__ wrow,
                                                int half /*lane>>4*/) {
    v16h b;
    const float* p = wrow + half * 16;
#pragma unroll
    for (int i = 0; i < 16; ++i) b[i] = (_Float16)p[i];
    return b;
}

__device__ __forceinline__ v16h load_b_from_f16(const _Float16* __restrict__ wrow,
                                                int half) {
    v16h b;
    const _Float16* p = wrow + half * 16;
#pragma unroll
    for (int i = 0; i < 16; ++i) b[i] = p[i];
    return b;
}

__global__ __launch_bounds__(1024)
void mlp_head_kernel(const float* __restrict__ pooled,
                     const float* __restrict__ W1, const float* __restrict__ b1,
                     const float* __restrict__ gamma, const float* __restrict__ beta,
                     const float* __restrict__ W2, const float* __restrict__ b2,
                     float* __restrict__ out) {
    __shared__ _Float16 sP[NBATCH * FEAT];   // pooled, f16        (4 KB)
    __shared__ float    sX[NBATCH * H1];     // fc1 output, f32    (32 KB)
    __shared__ _Float16 sH[NBATCH * H1];     // bn+relu, f16       (16 KB)

    const int tid  = threadIdx.x;
    const int wave = tid >> 5;
    const int lane = tid & 31;
    const int half = lane >> 4;
    const int mn   = lane & 15;

    // stage pooled into LDS as f16
    for (int i = tid; i < NBATCH * FEAT; i += 1024)
        sP[i] = (_Float16)pooled[i];
    __syncthreads();

    // ---------------- fc1: one 16x16 tile per wave (32 tiles) --------------
    {
        const int mt = wave & 1;        // 0..1  (rows 0..15 / 16..31)
        const int nt = wave >> 1;       // 0..15 (output cols)
        v8f acc = {};
#pragma unroll
        for (int kk = 0; kk < FEAT; kk += 32) {
            v16h a = load_a_f16(&sP[(mt * 16 + mn) * FEAT + kk], half);
            v16h b = load_b_from_f32(&W1[(nt * 16 + mn) * FEAT + kk], half);
            acc = __builtin_amdgcn_wmma_f32_16x16x32_f16(
                false, a, false, b, (short)0, acc, false, false);
        }
        const int n = nt * 16 + mn;
        const float bias = b1[n];
#pragma unroll
        for (int v = 0; v < 8; ++v) {
            const int m = mt * 16 + half * 8 + v;
            sX[m * H1 + n] = acc[v] + bias;
        }
    }
    __syncthreads();

    // ---------------- BatchNorm1d (training stats over 32 rows) + ReLU -----
    if (tid < H1) {
        float mean = 0.f;
#pragma unroll
        for (int r = 0; r < NBATCH; ++r) mean += sX[r * H1 + tid];
        mean *= (1.0f / NBATCH);
        float var = 0.f;
#pragma unroll
        for (int r = 0; r < NBATCH; ++r) {
            const float d = sX[r * H1 + tid] - mean;
            var += d * d;
        }
        var *= (1.0f / NBATCH);
        const float inv = rsqrtf(var + BN_EPS) * gamma[tid];
        const float bet = beta[tid];
#pragma unroll
        for (int r = 0; r < NBATCH; ++r) {
            float y = (sX[r * H1 + tid] - mean) * inv + bet;
            y = fmaxf(y, 0.0f);
            sH[r * H1 + tid] = (_Float16)y;
        }
    }
    __syncthreads();

    // ---------------- fc2: 16 tiles, waves 0..15, K=256 --------------------
    if (wave < 16) {
        const int mt = wave & 1;   // 0..1
        const int nt = wave >> 1;  // 0..7
        v8f acc = {};
#pragma unroll
        for (int kk = 0; kk < H1; kk += 32) {
            v16h a = load_a_f16(&sH[(mt * 16 + mn) * H1 + kk], half);
            v16h b = load_b_from_f32(&W2[(nt * 16 + mn) * H1 + kk], half);
            acc = __builtin_amdgcn_wmma_f32_16x16x32_f16(
                false, a, false, b, (short)0, acc, false, false);
        }
        const int n = nt * 16 + mn;
        const float bias = b2[n];
#pragma unroll
        for (int v = 0; v < 8; ++v) {
            const int m = mt * 16 + half * 8 + v;
            out[m * H2 + n] = acc[v] + bias;
        }
    }
}

// ---------------------------------------------------------------------------
// Launch
// ---------------------------------------------------------------------------
extern "C" void kernel_launch(void* const* d_in, const int* in_sizes, int n_in,
                              void* d_out, int out_size, void* d_ws, size_t ws_size,
                              hipStream_t stream) {
    const float* F     = (const float*)d_in[0];
    const int*   bidx  = (const int*)  d_in[1];
    // d_in[2] is B (scalar, = 32)
    const float* W1    = (const float*)d_in[3];
    const float* b1    = (const float*)d_in[4];
    const float* gamma = (const float*)d_in[5];
    const float* beta  = (const float*)d_in[6];
    const float* W2    = (const float*)d_in[7];
    const float* b2    = (const float*)d_in[8];

    const int N = in_sizes[0] / FEAT;   // 4194304
    float* pooled = (float*)d_ws;       // [32, 64] f32

    const int poolN = NBATCH * FEAT;    // 2048
    init_pooled_kernel<<<(poolN + 255) / 256, 256, 0, stream>>>(pooled, poolN);

    const int nBlocks = (N + ROWS_PER_BLOCK - 1) / ROWS_PER_BLOCK; // 2048
    segmax_kernel<<<nBlocks, 256, 0, stream>>>(F, bidx, pooled, N);

    mlp_head_kernel<<<1, 1024, 0, stream>>>(pooled, W1, b1, gamma, beta, W2, b2,
                                            (float*)d_out);
}
